// MultiHeadAttention_48069273977055
// MI455X (gfx1250) — matrix operand order repaired and checked by reading, not compiled, not run on hardware
//
#include <hip/hip_runtime.h>

// ---------------- problem constants ----------------
constexpr int kE = 768;          // embed
constexpr int kH = 12;           // heads
constexpr int kD = 64;           // head dim
constexpr int kS = 2048;         // seq
constexpr int kB = 2;            // batch
constexpr int kM = kB * kS;      // 4096 total rows
constexpr int WAVES = 8;         // waves per 256-thread GEMM block
constexpr int AW = 2;            // waves per attention block
#define NEG_BIG (-1.0e38f)

typedef __attribute__((ext_vector_type(16))) _Float16 v16h;
typedef __attribute__((ext_vector_type(8)))  _Float16 v8h;
typedef __attribute__((ext_vector_type(4)))  _Float16 v4h;
typedef __attribute__((ext_vector_type(8)))  float    v8f;
typedef __attribute__((ext_vector_type(4)))  unsigned int v4u;
typedef __attribute__((ext_vector_type(8)))  int v8i;
typedef __attribute__((ext_vector_type(4)))  int v4i;

union U16h { v16h v; v8h h8[2]; };

static __device__ __forceinline__ v8f zero8() {
  v8f z;
#pragma unroll
  for (int r = 0; r < 8; ++r) z[r] = 0.0f;
  return z;
}
static __device__ __forceinline__ float rmax16(float v) {
#pragma unroll
  for (int m = 1; m < 16; m <<= 1) v = fmaxf(v, __shfl_xor(v, m, 32));
  return v;
}
static __device__ __forceinline__ float rsum16(float v) {
#pragma unroll
  for (int m = 1; m < 16; m <<= 1) v += __shfl_xor(v, m, 32);
  return v;
}
static __device__ __forceinline__ U16h loadA(const _Float16* row, int kk, int half) {
  U16h a;
  a.h8[0] = *(const v8h*)(row + kk + 16 * half);
  a.h8[1] = *(const v8h*)(row + kk + 16 * half + 8);
  return a;
}
static __device__ __forceinline__ v16h loadB(const _Float16* brow, int t, int kk, int half) {
  return *(const v16h*)(brow + (size_t)t * 16 * kE + kk + 16 * half);
}

// ---------------- TDM staging: 2D tile global -> LDS ----------------
// rows x row_elems halves; gstride in elements. Done signal via TENSORcnt.
static __device__ __forceinline__ void stage_tile(const _Float16* g, _Float16* l,
                                                  unsigned row_elems, unsigned rows,
                                                  unsigned gstride, int lane) {
#if __has_builtin(__builtin_amdgcn_tensor_load_to_lds)
  (void)lane;
  const unsigned long long ga = (unsigned long long)(uintptr_t)g;
  v4u g0;
  g0.x = 1u;                                   // count=1, user descriptor
  g0.y = (unsigned)(uintptr_t)l;               // lds_addr (low 32 bits of flat LDS addr)
  g0.z = (unsigned)ga;                         // global_addr[31:0]
  g0.w = (unsigned)((ga >> 32) & 0x1FFFFFFu) | (2u << 30);   // addr[56:32] | type=2
  v8i g1;
  g1[0] = (int)(1u << 16);                                   // data_size=1 (2 bytes)
  g1[1] = (int)((row_elems & 0xFFFFu) << 16);                // tensor_dim0[15:0]
  g1[2] = (int)((row_elems >> 16) | ((rows & 0xFFFFu) << 16));   // dim0 hi | tensor_dim1 lo
  g1[3] = (int)((rows >> 16) | ((row_elems & 0xFFFFu) << 16));   // dim1 hi | tile_dim0
  g1[4] = (int)(rows & 0xFFFFu);                             // tile_dim1 (tile_dim2=0)
  g1[5] = (int)gstride;                                      // tensor_dim0_stride[31:0]
  g1[6] = 0;
  g1[7] = 0;
  const v4i z4 = {0, 0, 0, 0};
#if __clang_major__ >= 23
  const v8i z8 = {0, 0, 0, 0, 0, 0, 0, 0};
  __builtin_amdgcn_tensor_load_to_lds(g0, g1, z4, z4, z8, 0);
#else
  __builtin_amdgcn_tensor_load_to_lds(g0, g1, z4, z4, 0);
#endif
#else
  // fallback: cooperative per-lane copy (wave-private buffer)
  const int total = (int)(rows * row_elems);
  for (int i = lane * 8; i < total; i += 32 * 8) {
    const int r = i / (int)row_elems, c = i % (int)row_elems;
    *(v8h*)(l + i) = *(const v8h*)(g + (size_t)r * gstride + c);
  }
#endif
}
static __device__ __forceinline__ void wait_tiles_2() {
#if __has_builtin(__builtin_amdgcn_tensor_load_to_lds)
  __builtin_amdgcn_s_wait_tensorcnt(2);
#endif
}
static __device__ __forceinline__ void wait_tiles_0() {
#if __has_builtin(__builtin_amdgcn_tensor_load_to_lds)
  __builtin_amdgcn_s_wait_tensorcnt(0);
#endif
}

// ---------------- fp32 -> fp16 conversion ----------------
__global__ void __launch_bounds__(256)
cvt_f32_f16(const float* __restrict__ src, _Float16* __restrict__ dst, int n4) {
  int i = blockIdx.x * blockDim.x + threadIdx.x;
  if (i < n4) {
    float4 f = ((const float4*)src)[i];
    v4h h;
    h.x = (_Float16)f.x; h.y = (_Float16)f.y;
    h.z = (_Float16)f.z; h.w = (_Float16)f.w;
    ((v4h*)dst)[i] = h;
  }
}

// ---------------- QKV projection GEMM ----------------
// y[m,f] = sum_e x[m,e]*W[f,e]; one wave: 32x64 tile, ping-pong pipelined (K step 64).
__global__ void __launch_bounds__(256)
qkv_gemm(const _Float16* __restrict__ XH, const _Float16* __restrict__ WQKV,
         _Float16* __restrict__ QH, _Float16* __restrict__ KH, _Float16* __restrict__ VT,
         float* __restrict__ outK, float* __restrict__ outV) {
  const int wave = threadIdx.x >> 5;
  const int lane = threadIdx.x & 31;
  const int half = lane >> 4, ln = lane & 15;
  const int task = blockIdx.x * WAVES + wave;   // 0..4607
  const int mt = task / 36;                     // 32-row tile (0..127)
  const int ns = task % 36;                     // 64-feature strip over 2304
  const int mat = ns / kH;                      // 0=Q 1=K 2=V
  const int h   = ns % kH;

  const _Float16* arow0 = XH + (size_t)(mt * 32 + ln) * kE;
  const _Float16* arow1 = arow0 + (size_t)16 * kE;
  const _Float16* brow  = WQKV + (size_t)(ns * 64 + ln) * kE;

  v8f acc[2][4];
#pragma unroll
  for (int s = 0; s < 2; ++s)
#pragma unroll
    for (int t = 0; t < 4; ++t) acc[s][t] = zero8();

  U16h a0 = loadA(arow0, 0, half), a1 = loadA(arow1, 0, half);
  v16h bA[4];
#pragma unroll
  for (int t = 0; t < 4; ++t) bA[t] = loadB(brow, t, 0, half);

  for (int kk = 0; kk < kE; kk += 64) {         // ping-pong: two 32-wide sets per iter
    U16h c0 = loadA(arow0, kk + 32, half);      // set B fragments (kE % 64 == 0)
    U16h c1 = loadA(arow1, kk + 32, half);
    v16h bB[4];
#pragma unroll
    for (int t = 0; t < 4; ++t) bB[t] = loadB(brow, t, kk + 32, half);
    __builtin_prefetch(brow + kk + 128, 0, 1);  // global_prefetch_b8

#pragma unroll
    for (int t = 0; t < 4; ++t) {               // compute set A
      acc[0][t] = __builtin_amdgcn_wmma_f32_16x16x32_f16(false, a0.v, false, bA[t],
                                                         (short)0, acc[0][t], false, false);
      acc[1][t] = __builtin_amdgcn_wmma_f32_16x16x32_f16(false, a1.v, false, bA[t],
                                                         (short)0, acc[1][t], false, false);
    }

    if (kk + 64 < kE) {                         // reload set A in place for next iter
      a0 = loadA(arow0, kk + 64, half);
      a1 = loadA(arow1, kk + 64, half);
#pragma unroll
      for (int t = 0; t < 4; ++t) bA[t] = loadB(brow, t, kk + 64, half);
    }

#pragma unroll
    for (int t = 0; t < 4; ++t) {               // compute set B
      acc[0][t] = __builtin_amdgcn_wmma_f32_16x16x32_f16(false, c0.v, false, bB[t],
                                                         (short)0, acc[0][t], false, false);
      acc[1][t] = __builtin_amdgcn_wmma_f32_16x16x32_f16(false, c1.v, false, bB[t],
                                                         (short)0, acc[1][t], false, false);
    }
  }

#pragma unroll
  for (int sub = 0; sub < 2; ++sub) {
#pragma unroll
    for (int t = 0; t < 4; ++t) {
      const int d = t * 16 + ln;
#pragma unroll
      for (int r = 0; r < 8; ++r) {
        const int m = mt * 32 + sub * 16 + r + 8 * half;
        const int b = m >> 11;
        const int s = m & (kS - 1);
        const size_t bh = (size_t)(b * kH + h);
        const float val = acc[sub][t][r];
        if (mat == 0) {
          QH[(bh * kS + s) * kD + d] = (_Float16)val;
        } else if (mat == 1) {
          const size_t idx = (bh * kS + s) * kD + d;
          KH[idx] = (_Float16)val;
          outK[idx] = val;
        } else {
          VT[(bh * kD + d) * kS + s] = (_Float16)val;   // transposed for PV B-fragments
          outV[(bh * kS + s) * kD + d] = val;
        }
      }
    }
  }
}

// ---------------- causal flash attention ----------------
// One wave per (b,h,16-row q-tile); K/V tiles TDM-staged into LDS, double-buffered.
__global__ void __launch_bounds__(64)
flash_attn(const _Float16* __restrict__ QH, const _Float16* __restrict__ KH,
           const _Float16* __restrict__ VT, _Float16* __restrict__ AH) {
  __shared__ __align__(128) _Float16 Kb[AW][2][32][64];   // [wave][buf][key][d]
  __shared__ __align__(128) _Float16 Vb[AW][2][64][32];   // [wave][buf][d][key]
  __shared__ __align__(64)  _Float16 Plds[AW][16][32];
  const int wave = threadIdx.x >> 5;
  const int lane = threadIdx.x & 31;
  const int half = lane >> 4, ln = lane & 15;
  const int task = blockIdx.x * AW + wave;      // 0..3071
  const int qt = task & 127;                    // q tile index (S/16 = 128)
  const int bh = task >> 7;
  const int b = bh / kH, h = bh % kH;

  // Q A-fragments (16x64 as two 16x32 chunks)
  const _Float16* qp = QH + ((size_t)bh * kS + qt * 16 + ln) * kD;
  U16h qf[2];
#pragma unroll
  for (int c = 0; c < 2; ++c) {
    qf[c].h8[0] = *(const v8h*)(qp + c * 32 + 16 * half);
    qf[c].h8[1] = *(const v8h*)(qp + c * 32 + 16 * half + 8);
  }

  float mrun[8], lsum[8];
  v8f o[4];
#pragma unroll
  for (int r = 0; r < 8; ++r) { mrun[r] = NEG_BIG; lsum[r] = 0.0f; }
#pragma unroll
  for (int t = 0; t < 4; ++t) o[t] = zero8();

  const _Float16* kgbase = KH + (size_t)bh * kS * kD;
  const _Float16* vgbase = VT + (size_t)bh * kD * kS;
  const int nblk = (qt >> 1) + 1;               // 32-key blocks

  stage_tile(kgbase, &Kb[wave][0][0][0], 64, 32, 64, lane);
  stage_tile(vgbase, &Vb[wave][0][0][0], 32, 64, kS, lane);

  for (int blk = 0; blk < nblk; ++blk) {
    const int buf = blk & 1;
    if (blk + 1 < nblk) {                       // prefetch next 32-key block
      stage_tile(kgbase + (size_t)(blk + 1) * 32 * kD, &Kb[wave][buf ^ 1][0][0], 64, 32, 64, lane);
      stage_tile(vgbase + (blk + 1) * 32,              &Vb[wave][buf ^ 1][0][0], 32, 64, kS, lane);
      wait_tiles_2();                           // current buf's pair is complete
    } else {
      wait_tiles_0();
    }

    v8f sc[2];
#pragma unroll
    for (int sub = 0; sub < 2; ++sub) {
      const int kt = blk * 2 + sub;             // wave-uniform
      v8f s = zero8();
      if (kt <= qt) {
#pragma unroll
        for (int c = 0; c < 2; ++c) {
          const v16h bf = *(const v16h*)(&Kb[wave][buf][sub * 16 + ln][c * 32 + 16 * half]);
          s = __builtin_amdgcn_wmma_f32_16x16x32_f16(false, qf[c].v, false, bf,
                                                     (short)0, s, false, false);
        }
        if (kt == qt) {                         // causal mask on diagonal tile
#pragma unroll
          for (int r = 0; r < 8; ++r)
            if (ln > r + 8 * half) s[r] = NEG_BIG;
        }
      } else {
#pragma unroll
        for (int r = 0; r < 8; ++r) s[r] = NEG_BIG;
      }
      sc[sub] = s;
    }

    float mnew[8], scl[8];
#pragma unroll
    for (int r = 0; r < 8; ++r) {
      const float mx = rmax16(fmaxf(sc[0][r], sc[1][r]));
      mnew[r] = fmaxf(mrun[r], mx);
      scl[r] = __expf(mrun[r] - mnew[r]);
      mrun[r] = mnew[r];
    }
#pragma unroll
    for (int r = 0; r < 8; ++r) {
      const float p0 = __expf(sc[0][r] - mnew[r]);
      const float p1 = __expf(sc[1][r] - mnew[r]);
      Plds[wave][r + 8 * half][ln]      = (_Float16)p0;   // C-layout -> LDS
      Plds[wave][r + 8 * half][16 + ln] = (_Float16)p1;
      lsum[r] = lsum[r] * scl[r] + rsum16(p0 + p1);
#pragma unroll
      for (int t = 0; t < 4; ++t) o[t][r] *= scl[r];
    }

    asm volatile("s_wait_dscnt 0x0" ::: "memory");        // P store -> A-fragment reload

    U16h pf;
    pf.h8[0] = *(const v8h*)(&Plds[wave][ln][16 * half]);
    pf.h8[1] = *(const v8h*)(&Plds[wave][ln][16 * half + 8]);

#pragma unroll
    for (int t = 0; t < 4; ++t) {               // O += P * V
      const v16h vf = *(const v16h*)(&Vb[wave][buf][t * 16 + ln][16 * half]);
      o[t] = __builtin_amdgcn_wmma_f32_16x16x32_f16(false, pf.v, false, vf,
                                                    (short)0, o[t], false, false);
    }
  }

  const size_t obase = ((size_t)b * kS + qt * 16) * kE + h * kD;
#pragma unroll
  for (int r = 0; r < 8; ++r) {
    const float inv = 1.0f / lsum[r];
    const int m = r + 8 * half;
#pragma unroll
    for (int t = 0; t < 4; ++t)
      AH[obase + (size_t)m * kE + t * 16 + ln] = (_Float16)(o[t][r] * inv);
  }
}

// ---------------- output projection GEMM (+bias) ----------------
__global__ void __launch_bounds__(256)
out_gemm(const _Float16* __restrict__ AH, const _Float16* __restrict__ WOH,
         const float* __restrict__ bo, float* __restrict__ out) {
  const int wave = threadIdx.x >> 5;
  const int lane = threadIdx.x & 31;
  const int half = lane >> 4, ln = lane & 15;
  const int task = blockIdx.x * WAVES + wave;   // 0..1535
  const int mt = task / 12;                     // 32-row tile
  const int ns = task % 12;

  const _Float16* arow0 = AH + (size_t)(mt * 32 + ln) * kE;
  const _Float16* arow1 = arow0 + (size_t)16 * kE;
  const _Float16* brow  = WOH + (size_t)(ns * 64 + ln) * kE;

  v8f acc[2][4];
#pragma unroll
  for (int s = 0; s < 2; ++s)
#pragma unroll
    for (int t = 0; t < 4; ++t) acc[s][t] = zero8();

  U16h a0 = loadA(arow0, 0, half), a1 = loadA(arow1, 0, half);
  v16h bA[4];
#pragma unroll
  for (int t = 0; t < 4; ++t) bA[t] = loadB(brow, t, 0, half);

  for (int kk = 0; kk < kE; kk += 64) {
    U16h c0 = loadA(arow0, kk + 32, half);
    U16h c1 = loadA(arow1, kk + 32, half);
    v16h bB[4];
#pragma unroll
    for (int t = 0; t < 4; ++t) bB[t] = loadB(brow, t, kk + 32, half);
    __builtin_prefetch(brow + kk + 128, 0, 1);

#pragma unroll
    for (int t = 0; t < 4; ++t) {
      acc[0][t] = __builtin_amdgcn_wmma_f32_16x16x32_f16(false, a0.v, false, bA[t],
                                                         (short)0, acc[0][t], false, false);
      acc[1][t] = __builtin_amdgcn_wmma_f32_16x16x32_f16(false, a1.v, false, bA[t],
                                                         (short)0, acc[1][t], false, false);
    }

    if (kk + 64 < kE) {
      a0 = loadA(arow0, kk + 64, half);
      a1 = loadA(arow1, kk + 64, half);
#pragma unroll
      for (int t = 0; t < 4; ++t) bA[t] = loadB(brow, t, kk + 64, half);
    }

#pragma unroll
    for (int t = 0; t < 4; ++t) {
      acc[0][t] = __builtin_amdgcn_wmma_f32_16x16x32_f16(false, c0.v, false, bB[t],
                                                         (short)0, acc[0][t], false, false);
      acc[1][t] = __builtin_amdgcn_wmma_f32_16x16x32_f16(false, c1.v, false, bB[t],
                                                         (short)0, acc[1][t], false, false);
    }
  }

#pragma unroll
  for (int sub = 0; sub < 2; ++sub) {
#pragma unroll
    for (int t = 0; t < 4; ++t) {
      const int f = ns * 64 + t * 16 + ln;
      const float bias = bo[f];
#pragma unroll
      for (int r = 0; r < 8; ++r) {
        const int m = mt * 32 + sub * 16 + r + 8 * half;
        out[(size_t)m * kE + f] = acc[sub][t][r] + bias;
      }
    }
  }
}

// ---------------- host launcher ----------------
extern "C" void kernel_launch(void* const* d_in, const int* in_sizes, int n_in,
                              void* d_out, int out_size, void* d_ws, size_t ws_size,
                              hipStream_t stream) {
  (void)in_sizes; (void)n_in; (void)out_size; (void)ws_size;
  const float* x  = (const float*)d_in[0];
  const float* Wq = (const float*)d_in[1];
  const float* Wk = (const float*)d_in[2];
  const float* Wv = (const float*)d_in[3];
  const float* Wo = (const float*)d_in[4];
  const float* bo = (const float*)d_in[5];

  constexpr size_t NX  = (size_t)kM * kE;            // 3,145,728
  constexpr size_t NW  = (size_t)kE * kE;            //   589,824
  constexpr size_t NKV = (size_t)kB * kH * kS * kD;  // 3,145,728

  float* out  = (float*)d_out;
  float* outK = out + NX;
  float* outV = outK + NKV;

  _Float16* XH   = (_Float16*)d_ws;
  _Float16* WQKV = XH + NX;
  _Float16* WOH  = WQKV + 3 * NW;
  _Float16* QH   = WOH + NW;
  _Float16* KH   = QH + NKV;
  _Float16* VT   = KH + NKV;
  _Float16* AH   = VT + NKV;

  cvt_f32_f16<<<(int)(NX / 4 / 256), 256, 0, stream>>>(x, XH, (int)(NX / 4));
  cvt_f32_f16<<<(int)(NW / 4 / 256), 256, 0, stream>>>(Wq, WQKV, (int)(NW / 4));
  cvt_f32_f16<<<(int)(NW / 4 / 256), 256, 0, stream>>>(Wk, WQKV + NW, (int)(NW / 4));
  cvt_f32_f16<<<(int)(NW / 4 / 256), 256, 0, stream>>>(Wv, WQKV + 2 * NW, (int)(NW / 4));
  cvt_f32_f16<<<(int)(NW / 4 / 256), 256, 0, stream>>>(Wo, WOH, (int)(NW / 4));

  qkv_gemm<<<576, 256, 0, stream>>>(XH, WQKV, QH, KH, VT, outK, outV);
  flash_attn<<<1536, 64, 0, stream>>>(QH, KH, VT, AH);
  out_gemm<<<192, 256, 0, stream>>>(AH, WOH, bo, out);
}
